// Transformer_48301202211429
// MI455X (gfx1250) — matrix operand order repaired and checked by reading, not compiled, not run on hardware
//
#include <hip/hip_runtime.h>
#include <math.h>
#include <stdint.h>

// ---------------------------------------------------------------------------
// Problem constants (from the reference)
// ---------------------------------------------------------------------------
#define BB   16
#define TT   128
#define VVOC 64
#define EE   512
#define HH   8
#define DKK  64
#define SS   129          // T + 1
#define MM   (BB * SS)    // 2064 = 129 * 16  -> no M padding needed
#define NIT  (TT - 1)     // 127 scan iterations
#define LNEPS 1e-5f

typedef __attribute__((ext_vector_type(16))) __bf16 v16bf;
typedef __attribute__((ext_vector_type(8)))  float  v8f;

typedef __attribute__((ext_vector_type(4))) unsigned int tdm_u4;
typedef __attribute__((ext_vector_type(8))) int          tdm_i8;
typedef __attribute__((ext_vector_type(4))) int          tdm_i4;

__device__ __forceinline__ v8f wmma_bf16(v16bf a, v16bf b, v8f c) {
  return __builtin_amdgcn_wmma_f32_16x16x32_bf16(false, a, false, b, (short)0, c,
                                                 false, false);
}

__device__ __forceinline__ void pack_bf8(v16bf& f, int base, float4 a, float4 b) {
  f[base + 0] = (__bf16)a.x; f[base + 1] = (__bf16)a.y;
  f[base + 2] = (__bf16)a.z; f[base + 3] = (__bf16)a.w;
  f[base + 4] = (__bf16)b.x; f[base + 5] = (__bf16)b.y;
  f[base + 6] = (__bf16)b.z; f[base + 7] = (__bf16)b.w;
}

// ---------------------------------------------------------------------------
// Tensor Data Mover: 1-D contiguous copy of `nelem` bf16 elements to LDS.
// D# per CDNA5 ISA:
//   group0: count=1 | lds_addr | global_addr[56:0] | type=2
//   group1: data_size=1 (2B), tensor_dim0 = tile_dim0 = nelem, tile_dim1=0 (1-D)
// Completion tracked with TENSORcnt (s_wait_tensorcnt).
// ---------------------------------------------------------------------------
__device__ __forceinline__ void tdm_load_bf16_1d(unsigned lds_byte_off,
                                                 const __bf16* gsrc,
                                                 unsigned nelem) {
  unsigned long long ga = (unsigned long long)(uintptr_t)gsrc;
  tdm_u4 g0;
  g0[0] = 1u;                                                  // count=1
  g0[1] = lds_byte_off;                                        // LDS byte addr
  g0[2] = (unsigned)ga;                                        // gaddr[31:0]
  g0[3] = (unsigned)((ga >> 32) & 0x1FFFFFFu) | (2u << 30);    // gaddr[56:32]|type=2
  tdm_i8 g1;
  g1[0] = (int)(1u << 16);                   // data_size=1 (2 bytes), no multicast
  g1[1] = (int)((nelem & 0xFFFFu) << 16);    // tensor_dim0[15:0]
  g1[2] = (int)(nelem >> 16);                // tensor_dim0[31:16]; tensor_dim1=0
  g1[3] = (int)((nelem & 0xFFFFu) << 16);    // tile_dim0 = nelem (1-D tile)
  g1[4] = 0;                                 // tile_dim1 = tile_dim2 = 0 (unused)
  g1[5] = (int)nelem;                        // tensor_dim0_stride (unused for 1-D)
  g1[6] = 0;
  g1[7] = 0;
  tdm_i4 gz = {0, 0, 0, 0};
#if __clang_major__ >= 23
  tdm_i8 gz8 = {0, 0, 0, 0, 0, 0, 0, 0};
  __builtin_amdgcn_tensor_load_to_lds(g0, g1, gz, gz, gz8, 0);
#else
  __builtin_amdgcn_tensor_load_to_lds(g0, g1, gz, gz, 0);
#endif
}

__device__ __forceinline__ float block_reduce_sum(float v, float* red) {
  int tid = threadIdx.x;
  red[tid] = v;
  __syncthreads();
  for (int s = 128; s > 0; s >>= 1) {
    if (tid < s) red[tid] += red[tid + s];
    __syncthreads();
  }
  float r = red[0];
  __syncthreads();
  return r;
}

// ---------------------------------------------------------------------------
// Weight pack: f32 [K,N] row-major -> bf16 [(K/32)][N][32] so a B-fragment
// (lane: n = col, elems i: k = k0 + (lane>>4)*16 + i) is 16 contiguous bf16,
// and each (k-block, 512-col block) slab is one contiguous 32KB TDM tile.
// ---------------------------------------------------------------------------
__global__ void pack_kernel(const float* __restrict__ W, __bf16* __restrict__ Wp,
                            int K, int N) {
  int idx = blockIdx.x * 256 + threadIdx.x;
  if (idx >= K * N) return;
  int k = idx / N, n = idx % N;
  Wp[((size_t)(k >> 5) * N + n) * 32 + (k & 31)] = (__bf16)W[idx];
}

// rk7[c][n] = pos_emb[c] @ Wk + bk   (only 7 distinct relative-position keys)
__global__ void relkey_kernel(const float* __restrict__ pos_emb,
                              const float* __restrict__ Wk,
                              const float* __restrict__ bk,
                              float* __restrict__ rk7) {
  int c = blockIdx.x;
  for (int n = threadIdx.x; n < EE; n += blockDim.x) {
    float s = bk[n];
    for (int e = 0; e < EE; ++e) s += pos_emb[c * EE + e] * Wk[(size_t)e * EE + n];
    rk7[c * EE + n] = s;
  }
}

__global__ void init_kernel(float* hout, float* lanh, float* lah, int nout) {
  int i = blockIdx.x * 256 + threadIdx.x;
  if (i < nout) hout[i] = 0.0f;
  if (i < MM) { lanh[i] = 0.0f; lah[i] = -64.0f; }
}

// ---------------------------------------------------------------------------
// Embedding + initial LayerNorm:  h0 = LN(concat(x@We, eos), lno)
// ---------------------------------------------------------------------------
__global__ __launch_bounds__(256)
void embed_kernel(const float* __restrict__ x, const float* __restrict__ We,
                  const float* __restrict__ eos, const float* __restrict__ lns,
                  const float* __restrict__ lno, float* __restrict__ h) {
  __shared__ float xr[VVOC];
  __shared__ float er[EE];
  __shared__ float red[256];
  int row = blockIdx.x;               // b*S + s
  int b = row / SS, s = row % SS;
  int tid = threadIdx.x;
  if (s < TT && tid < VVOC) xr[tid] = x[((size_t)b * TT + s) * VVOC + tid];
  __syncthreads();
  float p = 0.f, p2 = 0.f;
  for (int e = tid; e < EE; e += 256) {
    float v;
    if (s == TT) v = eos[e];
    else { v = 0.f; for (int u = 0; u < VVOC; ++u) v += xr[u] * We[(size_t)u * EE + e]; }
    er[e] = v; p += v; p2 += v * v;
  }
  float sum = block_reduce_sum(p, red);
  float sq  = block_reduce_sum(p2, red);
  float mean = sum / EE;
  float inv = rsqrtf(sq / EE - mean * mean + LNEPS);
  for (int e = tid; e < EE; e += 256)
    h[(size_t)row * EE + e] = (er[e] - mean) * lns[e] * inv + lno[e];
}

// ---------------------------------------------------------------------------
// Generic bf16-WMMA GEMM with TDM-streamed weights:
//   C[M,N] = act(A[M,K] @ W + bias) (+ residual)
// Block covers 16 rows x 512 cols (8 waves, 16x64 strip each).
// Wave 0 double-buffers the contiguous 32KB per-k-step weight slab into LDS
// via tensor_load_to_lds; compute reads B-fragments with ds_load_b128.
// ACT: 0=none 1=gelu(tanh approx) 2=tanh
// ---------------------------------------------------------------------------
template <int ACT, bool RES>
__global__ __launch_bounds__(256)
void gemm_kernel(const float* __restrict__ A, const __bf16* __restrict__ Wp,
                 const float* __restrict__ bias, const float* __restrict__ res,
                 float* __restrict__ C, int N, int K) {
  __shared__ __bf16 wbuf[2][512 * 32];        // 2 x 32KB weight slabs
  const int lane = threadIdx.x & 31;
  const int wave = threadIdx.x >> 5;
  const int half = lane >> 4;
  const int l15  = lane & 15;
  const int mtile = blockIdx.x;
  const int nblk  = blockIdx.y;               // which 512-col block
  const int rowA  = mtile * 16 + l15;
  const int KB    = K >> 5;                   // k-steps of 32

  const v8f vz = {0.f, 0.f, 0.f, 0.f, 0.f, 0.f, 0.f, 0.f};
  v8f acc[4];
#pragma unroll
  for (int i = 0; i < 4; ++i) acc[i] = vz;

  const float* ap_base = A + (size_t)rowA * K;

  if (wave == 0) {                            // prologue: fetch slab 0
    tdm_load_bf16_1d((unsigned)(uintptr_t)&wbuf[0][0],
                     Wp + ((size_t)nblk * 512) * 32, 512 * 32);
  }

  for (int kb = 0; kb < KB; ++kb) {
    if (wave == 0) {
      if (kb + 1 < KB) {                      // async-prefetch next slab
        tdm_load_bf16_1d((unsigned)(uintptr_t)&wbuf[(kb + 1) & 1][0],
                         Wp + ((size_t)(kb + 1) * N + nblk * 512) * 32, 512 * 32);
        __builtin_amdgcn_s_wait_tensorcnt(1); // current slab has landed
      } else {
        __builtin_amdgcn_s_wait_tensorcnt(0);
      }
    }
    __syncthreads();                          // slab kb visible to all waves

    // A fragment: elems i (0..15) -> k0+half*16+i
    const float* ap = ap_base + kb * 32 + half * 16;
    __builtin_prefetch(ap + 32, 0, 1);        // next A k-block
    float4 a0 = *(const float4*)(ap);
    float4 a1 = *(const float4*)(ap + 4);
    float4 a2 = *(const float4*)(ap + 8);
    float4 a3 = *(const float4*)(ap + 12);
    v16bf af;
    pack_bf8(af, 0, a0, a1);
    pack_bf8(af, 8, a2, a3);

    const __bf16* wl = &wbuf[kb & 1][(wave * 64) * 32 + half * 16];
#pragma unroll
    for (int nt = 0; nt < 4; ++nt) {
      v16bf bf = *(const v16bf*)(wl + (nt * 16 + l15) * 32);
      acc[nt] = wmma_bf16(af, bf, acc[nt]);
    }
    __syncthreads();                          // done reading before overwrite
  }

  const int ncol0 = nblk * 512 + wave * 64;
#pragma unroll
  for (int nt = 0; nt < 4; ++nt) {
    int col = ncol0 + nt * 16 + l15;
    float bs = bias[col];
#pragma unroll
    for (int j = 0; j < 8; ++j) {
      int r = mtile * 16 + half * 8 + j;
      float v = acc[nt][j] + bs;
      if (ACT == 1) {
        float t = 0.7978845608f * (v + 0.044715f * v * v * v);
        v = 0.5f * v * (1.0f + tanhf(t));
      } else if (ACT == 2) {
        v = tanhf(v);
      }
      if (RES) v += res[(size_t)r * N + col];
      C[(size_t)r * N + col] = v;
    }
  }
}

// ---------------------------------------------------------------------------
// Fused relative-position attention for one (b, head).
// 9 waves (288 thr); wave w owns query row-tile w (rows w*16..w*16+15).
// scores = (q+cb)@k^T + r7[clip(t-T)] ; softmax in registers; a = p@v (WMMA).
// ---------------------------------------------------------------------------
__global__ __launch_bounds__(288)
void attn_kernel(const float* __restrict__ q, const float* __restrict__ k,
                 const float* __restrict__ v, const float* __restrict__ cb,
                 const float* __restrict__ rb, const float* __restrict__ rk7,
                 float* __restrict__ aout) {
  __shared__ __bf16 kS[144 * 64];      // k rows (T-padded to 144), row-major [T][d]
  __shared__ __bf16 vT[64 * 160];      // v transposed [d][T], T padded to 160
  __shared__ float  rkS[7 * 64];
  __shared__ float  cbS[64];
  __shared__ float  rbS[64];
  __shared__ float  r7S[9][16][8];     // per-wave: 7 rel dots per row (pad->8)
  __shared__ __bf16 pS[9][16 * 32];    // per-wave P-chunk staging (C -> A layout)

  const int bh = blockIdx.x;
  const int b = bh / HH, hh = bh % HH;
  const int tid = threadIdx.x;
  const int lane = tid & 31, wave = tid >> 5;
  const int half = lane >> 4, l15 = lane & 15;

  for (int idx = tid; idx < 144 * 64; idx += 288) {
    int T = idx >> 6, d = idx & 63;
    kS[idx] = (T < SS) ? (__bf16)k[((size_t)(b * SS + T)) * EE + hh * DKK + d]
                       : (__bf16)0.0f;
  }
  for (int idx = tid; idx < 64 * 160; idx += 288) {
    int d = idx / 160, T = idx % 160;
    vT[idx] = (T < SS) ? (__bf16)v[((size_t)(b * SS + T)) * EE + hh * DKK + d]
                       : (__bf16)0.0f;
  }
  if (tid < 7 * 64) rkS[tid] = rk7[(tid >> 6) * EE + hh * DKK + (tid & 63)];
  if (tid < 64) { cbS[tid] = cb[hh * DKK + tid]; rbS[tid] = rb[hh * DKK + tid]; }
  __syncthreads();

  const int ttile = wave;                    // 0..8
  const int trow = ttile * 16 + l15;
  const bool tvalid = trow < SS;
  const float* qp = q + ((size_t)(b * SS + (tvalid ? trow : 0))) * EE + hh * DKK;

  // A fragments of (q + content_bias): af0 covers d 0..31, af1 covers 32..63
  v16bf af0, af1;
#pragma unroll
  for (int i = 0; i < 8; ++i) {
    int d0 = half * 16 + i;
    int d1 = half * 16 + 8 + i;
    af0[i]     = (__bf16)(tvalid ? (qp[d0] + cbS[d0]) : 0.f);
    af0[i + 8] = (__bf16)(tvalid ? (qp[d1] + cbS[d1]) : 0.f);
    af1[i]     = (__bf16)(tvalid ? (qp[32 + d0] + cbS[32 + d0]) : 0.f);
    af1[i + 8] = (__bf16)(tvalid ? (qp[32 + d1] + cbS[32 + d1]) : 0.f);
  }

  // r7[row][c] = (q_row + rel_bias) . rk7[c]   (f32 VALU, tiny)
#pragma unroll
  for (int c = 0; c < 7; ++c) {
    float part = 0.f;
    for (int d = half * 32; d < half * 32 + 32; ++d) {
      float qv = tvalid ? qp[d] : 0.f;
      part += (qv + rbS[d]) * rkS[c * 64 + d];
    }
    part += __shfl_xor(part, 16, 32);
    r7S[wave][l15][c] = part;
  }
  __syncthreads();

  // ---- scores via WMMA: 9 column tiles of 16, K(d)=64 in two 32-chunks ----
  const v8f vz = {0.f, 0.f, 0.f, 0.f, 0.f, 0.f, 0.f, 0.f};
  v8f sc[9];
#pragma unroll
  for (int ct = 0; ct < 9; ++ct) sc[ct] = vz;
#pragma unroll
  for (int ct = 0; ct < 9; ++ct) {
    int Tc = ct * 16 + l15;
    v16bf b0 = *(const v16bf*)(kS + Tc * 64 + half * 16);        // d 0..31
    v16bf b1 = *(const v16bf*)(kS + Tc * 64 + 32 + half * 16);   // d 32..63
    sc[ct] = wmma_bf16(af0, b0, sc[ct]);
    sc[ct] = wmma_bf16(af1, b1, sc[ct]);
  }

  // add relative logits, scale, mask T >= S
  const float scale = 0.125f;                 // 1/sqrt(64)
#pragma unroll
  for (int ct = 0; ct < 9; ++ct) {
    int Tc = ct * 16 + l15;
#pragma unroll
    for (int j = 0; j < 8; ++j) {
      int t = ttile * 16 + half * 8 + j;
      int rel = t - Tc;
      rel = rel < -3 ? -3 : (rel > 3 ? 3 : rel);
      int cidx = rel < 0 ? rel + 7 : rel;     // pos_emb negative-index wrap
      float sv = (sc[ct][j] + r7S[wave][half * 8 + j][cidx]) * scale;
      if (Tc >= SS) sv = -1e30f;
      sc[ct][j] = sv;
    }
  }

  // ---- row softmax fully in registers (16-lane column groups) ----
#pragma unroll
  for (int j = 0; j < 8; ++j) {
    float mx = -1e30f;
#pragma unroll
    for (int ct = 0; ct < 9; ++ct) mx = fmaxf(mx, sc[ct][j]);
    for (int off = 1; off < 16; off <<= 1) mx = fmaxf(mx, __shfl_xor(mx, off, 32));
    float sum = 0.f;
#pragma unroll
    for (int ct = 0; ct < 9; ++ct) {
      float e = expf(sc[ct][j] - mx);
      sc[ct][j] = e; sum += e;
    }
    for (int off = 1; off < 16; off <<= 1) sum += __shfl_xor(sum, off, 32);
    float inv = 1.f / sum;
#pragma unroll
    for (int ct = 0; ct < 9; ++ct) sc[ct][j] *= inv;
  }

  // ---- a = p @ v : K(T) padded to 160, 5 chunks of 32; N(d)=64 -> 4 tiles --
  v8f oacc[4];
#pragma unroll
  for (int i = 0; i < 4; ++i) oacc[i] = vz;
  for (int kc = 0; kc < 5; ++kc) {
    // stage p chunk (tiles 2kc, 2kc+1) into LDS to re-layout C->A fragment
#pragma unroll
    for (int u = 0; u < 2; ++u) {
      int ct = 2 * kc + u;
#pragma unroll
      for (int j = 0; j < 8; ++j) {
        float pv = (ct < 9) ? sc[ct][j] : 0.f;
        pS[wave][(half * 8 + j) * 32 + u * 16 + l15] = (__bf16)pv;
      }
    }
    __syncthreads();
    v16bf paf;
    {
      const __bf16* pp = pS[wave] + l15 * 32;
#pragma unroll
      for (int i = 0; i < 8; ++i) {
        paf[i]     = pp[half * 16 + i];
        paf[i + 8] = pp[half * 16 + 8 + i];
      }
    }
#pragma unroll
    for (int dt = 0; dt < 4; ++dt) {
      const __bf16* vp = vT + (dt * 16 + l15) * 160 + kc * 32 + half * 16;
      v16bf bf = *(const v16bf*)vp;
      oacc[dt] = wmma_bf16(paf, bf, oacc[dt]);
    }
    __syncthreads();
  }

#pragma unroll
  for (int dt = 0; dt < 4; ++dt) {
#pragma unroll
    for (int j = 0; j < 8; ++j) {
      int t = ttile * 16 + half * 8 + j;
      if (t < SS)
        aout[((size_t)(b * SS + t)) * EE + hh * DKK + dt * 16 + l15] = oacc[dt][j];
    }
  }
}

// ---------------------------------------------------------------------------
// LayerNorm (one row per block, 512 elems)
// ---------------------------------------------------------------------------
__global__ __launch_bounds__(256)
void ln_kernel(const float* __restrict__ in, const float* __restrict__ s,
               const float* __restrict__ o, float* __restrict__ out) {
  __shared__ float red[256];
  int row = blockIdx.x, tid = threadIdx.x;
  const float* ip = in + (size_t)row * EE;
  float v0 = ip[tid], v1 = ip[tid + 256];
  float sum = block_reduce_sum(v0 + v1, red);
  float sq  = block_reduce_sum(v0 * v0 + v1 * v1, red);
  float mean = sum / EE;
  float inv = rsqrtf(sq / EE - mean * mean + LNEPS);
  out[(size_t)row * EE + tid]       = (v0 - mean) * s[tid] * inv + o[tid];
  out[(size_t)row * EE + tid + 256] = (v1 - mean) * s[tid + 256] * inv + o[tid + 256];
}

// ---------------------------------------------------------------------------
// ACT halting update (one row per block):
//   z = th . Wh2 + [2,-2]; log_softmax; log_halt = lanh + lg1; lanh += lg0;
//   lah = logaddexp(lah, log_halt); curr_n = LN(curr, lno);
//   h = halted*prev_h + (1-halted)*curr_n;  h_out += exp(log_halt)*curr_n
// ---------------------------------------------------------------------------
__global__ __launch_bounds__(256)
void update_kernel(const float* __restrict__ curr, const float* __restrict__ th,
                   const float* __restrict__ Wh2, const float* __restrict__ lns,
                   const float* __restrict__ lno, float* __restrict__ h,
                   float* __restrict__ lanh, float* __restrict__ lah,
                   float* __restrict__ hout) {
  __shared__ float red[256];
  int row = blockIdx.x, tid = threadIdx.x;
  const float* tp = th + (size_t)row * EE;
  const float* cp = curr + (size_t)row * EE;
  float t0 = tp[tid], t1 = tp[tid + 256];
  float c0 = cp[tid], c1 = cp[tid + 256];
  float z0 = block_reduce_sum(t0 * Wh2[tid * 2]     + t1 * Wh2[(tid + 256) * 2],     red);
  float z1 = block_reduce_sum(t0 * Wh2[tid * 2 + 1] + t1 * Wh2[(tid + 256) * 2 + 1], red);
  float sm = block_reduce_sum(c0 + c1, red);
  float sq = block_reduce_sum(c0 * c0 + c1 * c1, red);
  z0 += 2.0f; z1 -= 2.0f;
  float zm = fmaxf(z0, z1);
  float lse = zm + logf(expf(z0 - zm) + expf(z1 - zm));
  float lg0 = z0 - lse, lg1 = z1 - lse;
  float lanh_o = lanh[row], lah_o = lah[row];
  float log_halt = lanh_o + lg1;
  float halted = expf(lah_o);
  float p_halt = expf(log_halt);
  if (tid == 0) {
    lanh[row] = lanh_o + lg0;
    float m2 = fmaxf(lah_o, log_halt);
    lah[row] = m2 + logf(expf(lah_o - m2) + expf(log_halt - m2));
  }
  float mean = sm / EE;
  float inv = rsqrtf(sq / EE - mean * mean + LNEPS);
  size_t base = (size_t)row * EE;
  float cn0 = (c0 - mean) * lns[tid] * inv + lno[tid];
  float cn1 = (c1 - mean) * lns[tid + 256] * inv + lno[tid + 256];
  float hp0 = h[base + tid], hp1 = h[base + tid + 256];
  h[base + tid]       = halted * hp0 + (1.f - halted) * cn0;
  h[base + tid + 256] = halted * hp1 + (1.f - halted) * cn1;
  hout[base + tid]       += p_halt * cn0;
  hout[base + tid + 256] += p_halt * cn1;
}

// ---------------------------------------------------------------------------
// Host driver
// ---------------------------------------------------------------------------
extern "C" void kernel_launch(void* const* d_in, const int* in_sizes, int n_in,
                              void* d_out, int out_size, void* d_ws, size_t ws_size,
                              hipStream_t stream) {
  (void)in_sizes; (void)n_in; (void)ws_size;
  const float* x    = (const float*)d_in[0];
  const float* We   = (const float*)d_in[1];
  const float* eos  = (const float*)d_in[2];
  const float* Wq   = (const float*)d_in[3];
  const float* bq   = (const float*)d_in[4];
  const float* Wk   = (const float*)d_in[5];
  const float* bk   = (const float*)d_in[6];
  const float* Wv   = (const float*)d_in[7];
  const float* bv   = (const float*)d_in[8];
  const float* Wo   = (const float*)d_in[9];
  const float* bo   = (const float*)d_in[10];
  const float* cb   = (const float*)d_in[11];
  const float* rbb  = (const float*)d_in[12];
  const float* pos  = (const float*)d_in[13];
  const float* ln2s = (const float*)d_in[14];
  const float* ln2o = (const float*)d_in[15];
  const float* lnos = (const float*)d_in[16];
  const float* lnoo = (const float*)d_in[17];
  const float* W1   = (const float*)d_in[18];
  const float* b1   = (const float*)d_in[19];
  const float* W2   = (const float*)d_in[20];
  const float* b2   = (const float*)d_in[21];
  const float* Wh1  = (const float*)d_in[22];
  const float* bh1  = (const float*)d_in[23];
  const float* Wh2  = (const float*)d_in[24];
  float* hout = (float*)d_out;

  char* p = (char*)d_ws;
  auto alloc = [&](size_t bytes) -> char* {
    char* r = p;
    p += (bytes + 255) & ~(size_t)255;
    return r;
  };
  float* h    = (float*)alloc((size_t)MM * EE * 4);
  float* h1   = (float*)alloc((size_t)MM * EE * 4);
  float* hn   = (float*)alloc((size_t)MM * EE * 4);
  float* curr = (float*)alloc((size_t)MM * EE * 4);
  float* th   = (float*)alloc((size_t)MM * EE * 4);
  float* qb   = (float*)alloc((size_t)MM * EE * 4);
  float* kb   = (float*)alloc((size_t)MM * EE * 4);
  float* vb   = (float*)alloc((size_t)MM * EE * 4);
  float* ab   = (float*)alloc((size_t)MM * EE * 4);
  float* t1   = (float*)alloc((size_t)MM * 2 * EE * 4);
  float* lanh = (float*)alloc((size_t)MM * 4);
  float* lah  = (float*)alloc((size_t)MM * 4);
  float* rk7  = (float*)alloc((size_t)7 * EE * 4);
  __bf16* Wqp  = (__bf16*)alloc((size_t)EE * EE * 2);
  __bf16* Wkp  = (__bf16*)alloc((size_t)EE * EE * 2);
  __bf16* Wvp  = (__bf16*)alloc((size_t)EE * EE * 2);
  __bf16* Wop  = (__bf16*)alloc((size_t)EE * EE * 2);
  __bf16* Wh1p = (__bf16*)alloc((size_t)EE * EE * 2);
  __bf16* W1p  = (__bf16*)alloc((size_t)EE * 2 * EE * 2);
  __bf16* W2p  = (__bf16*)alloc((size_t)2 * EE * EE * 2);

  // ---- one-time prep ----
  init_kernel<<<(out_size + 255) / 256, 256, 0, stream>>>(hout, lanh, lah, out_size);
  {
    int n1 = EE * EE, n2 = EE * 2 * EE;
    pack_kernel<<<(n1 + 255) / 256, 256, 0, stream>>>(Wq,  Wqp,  EE, EE);
    pack_kernel<<<(n1 + 255) / 256, 256, 0, stream>>>(Wk,  Wkp,  EE, EE);
    pack_kernel<<<(n1 + 255) / 256, 256, 0, stream>>>(Wv,  Wvp,  EE, EE);
    pack_kernel<<<(n1 + 255) / 256, 256, 0, stream>>>(Wo,  Wop,  EE, EE);
    pack_kernel<<<(n1 + 255) / 256, 256, 0, stream>>>(Wh1, Wh1p, EE, EE);
    pack_kernel<<<(n2 + 255) / 256, 256, 0, stream>>>(W1,  W1p,  EE, 2 * EE);
    pack_kernel<<<(n2 + 255) / 256, 256, 0, stream>>>(W2,  W2p,  2 * EE, EE);
  }
  relkey_kernel<<<7, 256, 0, stream>>>(pos, Wk, bk, rk7);
  embed_kernel<<<MM, 256, 0, stream>>>(x, We, eos, lnos, lnoo, h);

  dim3 g512(MM / 16, 1), g1024(MM / 16, 2);
  for (int it = 0; it < NIT; ++it) {
    gemm_kernel<0, false><<<g512, 256, 0, stream>>>(h, Wqp, bq, nullptr, qb, EE, EE);
    gemm_kernel<0, false><<<g512, 256, 0, stream>>>(h, Wkp, bk, nullptr, kb, EE, EE);
    gemm_kernel<0, false><<<g512, 256, 0, stream>>>(h, Wvp, bv, nullptr, vb, EE, EE);
    attn_kernel<<<BB * HH, 288, 0, stream>>>(qb, kb, vb, cb, rbb, rk7, ab);
    gemm_kernel<0, true ><<<g512, 256, 0, stream>>>(ab, Wop, bo, h, h1, EE, EE);
    ln_kernel<<<MM, 256, 0, stream>>>(h1, ln2s, ln2o, hn);
    gemm_kernel<1, false><<<g1024, 256, 0, stream>>>(hn, W1p, b1, nullptr, t1, 2 * EE, EE);
    gemm_kernel<0, true ><<<g512, 256, 0, stream>>>(t1, W2p, b2, h1, curr, EE, 2 * EE);
    gemm_kernel<2, false><<<g512, 256, 0, stream>>>(curr, Wh1p, bh1, nullptr, th, EE, EE);
    update_kernel<<<MM, 256, 0, stream>>>(curr, th, Wh2, lnos, lnoo,
                                          h, lanh, lah, hout);
  }
}
